// BoundaryConvLayer_20315195310328
// MI455X (gfx1250) — matrix operand order repaired and checked by reading, not compiled, not run on hardware
//
#include <hip/hip_runtime.h>
#include <hip/hip_bf16.h>
#include <math.h>

// ---------------------------------------------------------------------------
// BoundaryConvLayer for MI455X (gfx1250), wave32 + WMMA bf16.
// ---------------------------------------------------------------------------

#define TPB 256
static constexpr int DIM = 128;   // feature dim
static constexpr int ASR = 136;   // activation stage stride (bf16 elems), 272B rows -> 16B aligned chunks
static constexpr int WSR = 144;   // weight stage stride (bf16 elems), 288B rows -> 32B aligned chunks

typedef __attribute__((ext_vector_type(16))) __bf16 v16bf;
typedef __attribute__((ext_vector_type(8)))  __bf16 v8bf;
typedef __attribute__((ext_vector_type(8)))  float  v8f;

// Branchless tanh-approx gelu using native v_exp_f32 / v_rcp_f32.
// tanh(u) = 1 - 2s/(1+s), s = exp(-2u) = exp2(-2*log2e*u)
__device__ __forceinline__ float gelu_tanh(float x) {
    const float k0 = 0.7978845608028654f;  // sqrt(2/pi)
    const float k1 = 0.044715f;
    float u = k0 * fmaf(k1 * x, x * x, x);
    float t = fminf(fmaxf(u, -15.0f), 15.0f);          // med3, avoids exp overflow
    float s = __builtin_amdgcn_exp2f(t * -2.8853900817779268f); // -2/ln(2)
    float r = __builtin_amdgcn_rcpf(1.0f + s);
    float th = 1.0f - 2.0f * s * r;
    return 0.5f * x * (1.0f + th);
}

// ---------------------------------------------------------------------------
// Fused two-layer MLP over a 128-row block per workgroup, computed with
// v_wmma_f32_16x16x32_bf16. mode: 0/1 = branch (LayerNorm epilogue),
// 2 = final head (h assembled from aggregation buffers, +b2+gamma epilogue).
// ---------------------------------------------------------------------------
__global__ __launch_bounds__(TPB, 1)
void fused_mlp_kernel(const float* __restrict__ inx,
                      const float* __restrict__ w1, const float* __restrict__ b1,
                      const float* __restrict__ w2, const float* __restrict__ b2,
                      const float* __restrict__ gvec, const float* __restrict__ bevec,
                      const float* __restrict__ agg1, const float* __restrict__ agg2,
                      const float* __restrict__ ratep,
                      const float* __restrict__ ind_bd, const float* __restrict__ inv_sq,
                      const float* __restrict__ gammap,
                      float* __restrict__ outp,
                      int Nn, int mode)
{
    __shared__ __bf16 wt1[DIM * WSR];  // w1 transposed: wt1[n*WSR + k]
    __shared__ __bf16 wt2[DIM * WSR];  // w2 transposed
    __shared__ __bf16 xs [DIM * ASR];  // input rows (bf16), [r*ASR + c]
    __shared__ __bf16 ms [DIM * ASR];  // mid (gelu) rows (bf16)

    const int tid  = threadIdx.x;
    const int wave = tid >> 5;
    const int lane = tid & 31;
    const int half = lane >> 4;   // 0: lanes 0-15, 1: lanes 16-31
    const int lr   = lane & 15;
    const int m0   = blockIdx.x * 128;

    // ---- stage both weight matrices, transposed, as bf16 ----
    for (int i = tid; i < DIM * DIM; i += TPB) {
        int k = i >> 7, n = i & (DIM - 1);
        wt1[n * WSR + k] = (__bf16)w1[i];
        wt2[n * WSR + k] = (__bf16)w2[i];
    }
    // ---- stage 128 input rows as bf16 (mode 2 builds h on the fly) ----
    for (int i = tid; i < DIM * DIM; i += TPB) {
        int r = i >> 7, c = i & (DIM - 1);
        int node = m0 + r;
        float v = 0.0f;
        if (node < Nn) {
            if (mode < 2) {
                v = inx[(size_t)node * DIM + c];
            } else {
                float ib = ind_bd[node];
                float ii = 1.0f - ib;
                float w  = inv_sq[node];
                v = ii * w * agg1[(size_t)node * DIM + c]
                  + ratep[(size_t)node * DIM + c] * ib * w * agg2[(size_t)node * DIM + c];
            }
        }
        xs[r * ASR + c] = (__bf16)v;
    }
    __syncthreads();

    // =======================================================================
    // GEMM 1: mid = gelu(x @ w1 + b1)
// A layout (bf16 16x32): lanes 0-15 row M=lr hold K kb*32+0..15,
// lanes 16-31 same rows hold K kb*32+16..31  -> two 16B LDS chunks.
    // =======================================================================
    v16bf a1[4];
    {
        const __bf16* arow = &xs[(wave * 16 + lr) * ASR];
        #pragma unroll
        for (int kb = 0; kb < 4; ++kb) {
            v8bf lo = *(const v8bf*)(arow + kb * 32 + half * 16);
            v8bf hi = *(const v8bf*)(arow + kb * 32 + half * 16 + 8);
            v16bf t;
            #pragma unroll
            for (int j = 0; j < 8; ++j) { t[j] = lo[j]; t[j + 8] = hi[j]; }
            a1[kb] = t;
        }
    }

    #pragma unroll
    for (int nb = 0; nb < 8; ++nb) {
        v8f acc = {0.f, 0.f, 0.f, 0.f, 0.f, 0.f, 0.f, 0.f};
        // B layout (bf16 32x16): lanes 0-15 col N=lr hold K kb*32+0..15,
        // lanes 16-31 hold K kb*32+16..31 -> one 32B contiguous LDS chunk.
        const __bf16* bcol = &wt1[(nb * 16 + lr) * WSR + half * 16];
        #pragma unroll
        for (int kb = 0; kb < 4; ++kb) {
            v16bf b = *(const v16bf*)(bcol + kb * 32);
            acc = __builtin_amdgcn_wmma_f32_16x16x32_bf16(false, a1[kb], false, b,
                                                          (short)0, acc, false, false);
        }
        int n = nb * 16 + lr;
        float bias = b1[n];
        // D layout: VGPR v holds row M = v + 8*half, col N = lr
        #pragma unroll
        for (int v = 0; v < 8; ++v) {
            int m = v + 8 * half;
            float g = gelu_tanh(acc[v] + bias);
            ms[(wave * 16 + m) * ASR + n] = (__bf16)g;
        }
    }
    __syncthreads();

    // =======================================================================
    // GEMM 2: out = mid @ w2 + b2, then LN (mode<2) or +gamma (mode 2)
    // =======================================================================
    v16bf a2[4];
    {
        const __bf16* arow = &ms[(wave * 16 + lr) * ASR];
        #pragma unroll
        for (int kb = 0; kb < 4; ++kb) {
            v8bf lo = *(const v8bf*)(arow + kb * 32 + half * 16);
            v8bf hi = *(const v8bf*)(arow + kb * 32 + half * 16 + 8);
            v16bf t;
            #pragma unroll
            for (int j = 0; j < 8; ++j) { t[j] = lo[j]; t[j + 8] = hi[j]; }
            a2[kb] = t;
        }
    }

    float vals[64];  // [nb*8 + v]
    #pragma unroll
    for (int nb = 0; nb < 8; ++nb) {
        v8f acc = {0.f, 0.f, 0.f, 0.f, 0.f, 0.f, 0.f, 0.f};
        const __bf16* bcol = &wt2[(nb * 16 + lr) * WSR + half * 16];
        #pragma unroll
        for (int kb = 0; kb < 4; ++kb) {
            v16bf b = *(const v16bf*)(bcol + kb * 32);
            acc = __builtin_amdgcn_wmma_f32_16x16x32_bf16(false, a2[kb], false, b,
                                                          (short)0, acc, false, false);
        }
        float bias = b2[nb * 16 + lr];
        #pragma unroll
        for (int v = 0; v < 8; ++v) vals[nb * 8 + v] = acc[v] + bias;
    }

    if (mode < 2) {
        // LayerNorm across each row directly in the D-register layout:
        // row M = v + 8*half lives in the 16 lanes of this half, 8 nb tiles.
        #pragma unroll
        for (int v = 0; v < 8; ++v) {
            float rs = 0.f, ss = 0.f;
            #pragma unroll
            for (int nb = 0; nb < 8; ++nb) { float x = vals[nb * 8 + v]; rs += x; ss += x * x; }
            #pragma unroll
            for (int m = 1; m < 16; m <<= 1) {  // stays within the 16-lane half
                rs += __shfl_xor(rs, m, 32);
                ss += __shfl_xor(ss, m, 32);
            }
            float mean = rs * (1.0f / 128.0f);
            float var  = ss * (1.0f / 128.0f) - mean * mean;
            float rstd = __builtin_amdgcn_rsqf(var + 1e-5f);
            int row = m0 + wave * 16 + v + 8 * half;
            if (row < Nn) {
                #pragma unroll
                for (int nb = 0; nb < 8; ++nb) {
                    int n = nb * 16 + lr;
                    outp[(size_t)row * DIM + n] =
                        (vals[nb * 8 + v] - mean) * rstd * gvec[n] + bevec[n];
                }
            }
        }
    } else {
        #pragma unroll
        for (int v = 0; v < 8; ++v) {
            int row = m0 + wave * 16 + v + 8 * half;
            if (row < Nn) {
                #pragma unroll
                for (int nb = 0; nb < 8; ++nb) {
                    int n = nb * 16 + lr;
                    outp[(size_t)row * DIM + n] =
                        vals[nb * 8 + v] + gammap[(size_t)row * DIM + n];
                }
            }
        }
    }
}

// ---------------------------------------------------------------------------
// Edge / node kernels (memory-bound side)
// ---------------------------------------------------------------------------
__global__ void deg_kernel(const int* __restrict__ src, const int* __restrict__ dst,
                           const float* __restrict__ ind_bd,
                           float* __restrict__ deg_full, float* __restrict__ deg_int, int E_)
{
    int e = blockIdx.x * blockDim.x + threadIdx.x;
    if (e >= E_) return;
    int s = src[e], d = dst[e];
    atomicAdd(&deg_full[d], 1.0f);
    atomicAdd(&deg_int[d], 1.0f - ind_bd[s]);
}

__global__ void invsq_kernel(const float* __restrict__ ind_bd,
                             const float* __restrict__ deg_full,
                             const float* __restrict__ deg_int,
                             float* __restrict__ inv_sq, int Nn)
{
    int n = blockIdx.x * blockDim.x + threadIdx.x;
    if (n >= Nn) return;
    float ib = ind_bd[n];
    float ii = 1.0f - ib;
    float p  = ib * deg_full[n] + (ii - ib) * deg_int[n] + 1.0f;
    inv_sq[n] = 1.0f / sqrtf(p);
}

// One 32-lane wave per edge; each lane carries a float4 (4 of 128 features).
__global__ void msg_kernel(const int* __restrict__ src, const int* __restrict__ dst,
                           const float* __restrict__ ind_bd, const float* __restrict__ inv_sq,
                           const float* __restrict__ xt,
                           float* __restrict__ agg1, float* __restrict__ agg2, int E_)
{
    int g    = blockIdx.x * blockDim.x + threadIdx.x;
    int e    = g >> 5;
    int lane = g & 31;
    if (e >= E_) return;
    int s = src[e], d = dst[e];
    float ii = 1.0f - ind_bd[s];
    float w  = inv_sq[s];
    const float4 x4 = *(const float4*)(&xt[(size_t)s * DIM + lane * 4]);
    float bx = x4.x * w, by = x4.y * w, bz = x4.z * w, bw = x4.w * w;
    float m2x = bx * ii, m2y = by * ii, m2z = bz * ii, m2w = bw * ii;
    float* a1 = &agg1[(size_t)d * DIM + lane * 4];
    float* a2 = &agg2[(size_t)d * DIM + lane * 4];
    atomicAdd(a1 + 0, bx + m2x); atomicAdd(a1 + 1, by + m2y);
    atomicAdd(a1 + 2, bz + m2z); atomicAdd(a1 + 3, bw + m2w);
    atomicAdd(a2 + 0, m2x);      atomicAdd(a2 + 1, m2y);
    atomicAdd(a2 + 2, m2z);      atomicAdd(a2 + 3, m2w);
}

// ---------------------------------------------------------------------------
extern "C" void kernel_launch(void* const* d_in, const int* in_sizes, int n_in,
                              void* d_out, int out_size, void* d_ws, size_t ws_size,
                              hipStream_t stream)
{
    const float* xt     = (const float*)d_in[0];
    const float* x0     = (const float*)d_in[1];
    const float* ind_bd = (const float*)d_in[2];
    const int*   eidx   = (const int*)  d_in[3];
    const float* w_r1 = (const float*)d_in[4];  const float* b_r1 = (const float*)d_in[5];
    const float* w_r2 = (const float*)d_in[6];  const float* b_r2 = (const float*)d_in[7];
    const float* g_r  = (const float*)d_in[8];  const float* be_r = (const float*)d_in[9];
    const float* w_g1 = (const float*)d_in[10]; const float* b_g1 = (const float*)d_in[11];
    const float* w_g2 = (const float*)d_in[12]; const float* b_g2 = (const float*)d_in[13];
    const float* g_g  = (const float*)d_in[14]; const float* be_g = (const float*)d_in[15];
    const float* w_f1 = (const float*)d_in[16]; const float* b_f1 = (const float*)d_in[17];
    const float* w_f2 = (const float*)d_in[18]; const float* b_f2 = (const float*)d_in[19];
    float* out = (float*)d_out;

    const int Nn = in_sizes[0] / DIM;
    const int E_ = in_sizes[3] / 2;
    const int* src = eidx;
    const int* dst = eidx + E_;

    // workspace layout (floats). Zeroed region first.
    float* ws = (float*)d_ws;
    size_t o = 0;
    float* deg_full = ws + o; o += (size_t)Nn;
    float* deg_int  = ws + o; o += (size_t)Nn;
    float* agg1     = ws + o; o += (size_t)Nn * DIM;
    float* agg2     = ws + o; o += (size_t)Nn * DIM;
    size_t zeroFloats = o;
    float* inv_sq   = ws + o; o += (size_t)Nn;
    float* rate     = ws + o; o += (size_t)Nn * DIM;
    float* gamma    = ws + o; o += (size_t)Nn * DIM;

    hipMemsetAsync(d_ws, 0, zeroFloats * sizeof(float), stream);

    const int gBlocks = (Nn + 127) / 128;

    // rate = LN(gelu(xt @ w_r1 + b_r1) @ w_r2 + b_r2)
    fused_mlp_kernel<<<gBlocks, TPB, 0, stream>>>(
        xt, w_r1, b_r1, w_r2, b_r2, g_r, be_r,
        agg1, agg2, rate, ind_bd, inv_sq, gamma, rate, Nn, 0);

    // gamma = LN(gelu(x0 @ w_g1 + b_g1) @ w_g2 + b_g2)
    fused_mlp_kernel<<<gBlocks, TPB, 0, stream>>>(
        x0, w_g1, b_g1, w_g2, b_g2, g_g, be_g,
        agg1, agg2, rate, ind_bd, inv_sq, gamma, gamma, Nn, 1);

    deg_kernel<<<(E_ + TPB - 1) / TPB, TPB, 0, stream>>>(src, dst, ind_bd, deg_full, deg_int, E_);
    invsq_kernel<<<(Nn + TPB - 1) / TPB, TPB, 0, stream>>>(ind_bd, deg_full, deg_int, inv_sq, Nn);

    long long msgThreads = (long long)E_ * 32;
    msg_kernel<<<(unsigned)((msgThreads + TPB - 1) / TPB), TPB, 0, stream>>>(
        src, dst, ind_bd, inv_sq, xt, agg1, agg2, E_);

    // out = gelu(h @ w_f1 + b_f1) @ w_f2 + b_f2 + gamma   (h built in-kernel)
    fused_mlp_kernel<<<gBlocks, TPB, 0, stream>>>(
        xt, w_f1, b_f1, w_f2, b_f2, b_f2, b_f2,
        agg1, agg2, rate, ind_bd, inv_sq, gamma, out, Nn, 2);
}
